// TemporalSSMLayer_12378095747610
// MI455X (gfx1250) — matrix operand order repaired and checked by reading, not compiled, not run on hardware
//
#include <hip/hip_runtime.h>
#include <hip/hip_bf16.h>

#define N_TOK 8192
#define DMODEL 1024
#define SDIM 16
#define EPSLN 1e-5f
#define KSTEPS (DMODEL / 32)

typedef __attribute__((ext_vector_type(16))) __bf16 v16bf;
typedef __attribute__((ext_vector_type(8)))  float  v8f;
typedef __attribute__((ext_vector_type(4)))  float  f4;

union Frag { unsigned u[8]; v16bf v; };

__device__ __forceinline__ unsigned pack2bf(float a, float b) {
    union { float f; unsigned u; } x, y;
    x.f = a; y.f = b;
    unsigned ra = (x.u + 0x7FFFu + ((x.u >> 16) & 1u)) >> 16;  // RNE
    unsigned rb = (y.u + 0x7FFFu + ((y.u >> 16) & 1u)) >> 16;
    return (ra & 0xFFFFu) | (rb << 16);
}

// out[n,j] = epi( sum_k x[n,k] * W[j,k] + bias[j] )
// blockIdx.z == 0 : W=Wd, bias=bd, out=delta, epi=softplus
// blockIdx.z == 1 : W=Wg, bias=bg, out=gate,  epi=silu
__global__ __launch_bounds__(256)
void ssm_gemm_epi_kernel(const float* __restrict__ x,
                         const float* __restrict__ Wd, const float* __restrict__ bd,
                         const float* __restrict__ Wg, const float* __restrict__ bg,
                         float* __restrict__ delta_o, float* __restrict__ gate_o)
{
    const int which = blockIdx.z;
    const float* W    = which ? Wg : Wd;
    const float* bias = which ? bg : bd;
    float*       out  = which ? gate_o : delta_o;

    // double-buffered tiles: bf16 pairs, padded stride (17 coprime with 64 banks)
    __shared__ unsigned As[2][128][17];
    __shared__ unsigned Bs[2][128][17];

    const int tid  = threadIdx.x;
    const int lane = tid & 31;
    const int wave = tid >> 5;
    const int wm   = wave & 3;     // 4 M-groups of 32 rows
    const int wn   = wave >> 2;    // 2 N-groups of 64 cols
    const int half = lane >> 4;    // lane group (K striping)
    const int l16  = lane & 15;

    const int row0 = blockIdx.x * 128;   // token rows
    const int col0 = blockIdx.y * 128;   // output feature cols

    // per-thread staging slots: slot = tid + it*256 -> (row, 4-col group)
    int srow[4], sc4[4];
#pragma unroll
    for (int it = 0; it < 4; ++it) {
        int slot = tid + it * 256;       // 0..1023
        srow[it] = slot >> 3;
        sc4[it]  = slot & 7;
    }

    v8f acc[2][4];
#pragma unroll
    for (int i = 0; i < 2; ++i)
#pragma unroll
        for (int j = 0; j < 4; ++j) acc[i][j] = (v8f)0.0f;

    float4 xr[4], wr[4];

    // prologue: fetch K-tile 0 and stage into buffer 0
#pragma unroll
    for (int it = 0; it < 4; ++it) {
        xr[it] = *(const float4*)&x[(size_t)(row0 + srow[it]) * DMODEL + sc4[it] * 4];
        wr[it] = *(const float4*)&W[(size_t)(col0 + srow[it]) * DMODEL + sc4[it] * 4];
    }
#pragma unroll
    for (int it = 0; it < 4; ++it) {
        As[0][srow[it]][sc4[it] * 2 + 0] = pack2bf(xr[it].x, xr[it].y);
        As[0][srow[it]][sc4[it] * 2 + 1] = pack2bf(xr[it].z, xr[it].w);
        Bs[0][srow[it]][sc4[it] * 2 + 0] = pack2bf(wr[it].x, wr[it].y);
        Bs[0][srow[it]][sc4[it] * 2 + 1] = pack2bf(wr[it].z, wr[it].w);
    }

    for (int kt = 0; kt < KSTEPS; ++kt) {
        __syncthreads();                 // buffer (kt&1) ready; prev reads done
        const int cur = kt & 1;

        // prefetch next K-tile from global while WMMAs run on current buffer
        if (kt + 1 < KSTEPS) {
            const int k0 = (kt + 1) * 32;
#pragma unroll
            for (int it = 0; it < 4; ++it) {
                xr[it] = *(const float4*)&x[(size_t)(row0 + srow[it]) * DMODEL + k0 + sc4[it] * 4];
                wr[it] = *(const float4*)&W[(size_t)(col0 + srow[it]) * DMODEL + k0 + sc4[it] * 4];
            }
        }

        // Gather fragments per documented 16-bit WMMA layouts (K pairs per VGPR)
        Frag a[2], b[4];
#pragma unroll
        for (int mi = 0; mi < 2; ++mi) {
            int mr = wm * 32 + mi * 16 + l16;
#pragma unroll
            for (int v = 0; v < 4; ++v) a[mi].u[v]     = As[cur][mr][4 * half + v];     // K 0-7 / 8-15
#pragma unroll
            for (int v = 0; v < 4; ++v) a[mi].u[4 + v] = As[cur][mr][8 + 4 * half + v]; // K 16-23 / 24-31
        }
#pragma unroll
        for (int ni = 0; ni < 4; ++ni) {
            int nc = wn * 64 + ni * 16 + l16;
#pragma unroll
            for (int v = 0; v < 8; ++v) b[ni].u[v] = Bs[cur][nc][4 * half + v + 4 * (v >> 2)]; // K 0-7,16-23 / 8-15,24-31
        }

#pragma unroll
        for (int mi = 0; mi < 2; ++mi)
#pragma unroll
            for (int ni = 0; ni < 4; ++ni)
                acc[mi][ni] = __builtin_amdgcn_wmma_f32_16x16x32_bf16(
                    false, a[mi].v, false, b[ni].v, (short)0, acc[mi][ni], false, false);

        // stage prefetched tile into the other buffer (no race: disjoint buffer,
        // next-iteration barrier orders these stores before their readers)
        if (kt + 1 < KSTEPS) {
            const int nxt = cur ^ 1;
#pragma unroll
            for (int it = 0; it < 4; ++it) {
                As[nxt][srow[it]][sc4[it] * 2 + 0] = pack2bf(xr[it].x, xr[it].y);
                As[nxt][srow[it]][sc4[it] * 2 + 1] = pack2bf(xr[it].z, xr[it].w);
                Bs[nxt][srow[it]][sc4[it] * 2 + 0] = pack2bf(wr[it].x, wr[it].y);
                Bs[nxt][srow[it]][sc4[it] * 2 + 1] = pack2bf(wr[it].z, wr[it].w);
            }
        }
    }

    // Epilogue: bias + softplus / silu, scatter per C/D layout
#pragma unroll
    for (int mi = 0; mi < 2; ++mi) {
#pragma unroll
        for (int ni = 0; ni < 4; ++ni) {
            int col = col0 + wn * 64 + ni * 16 + l16;
            float bv = bias[col];
#pragma unroll
            for (int r = 0; r < 8; ++r) {
                int row = row0 + wm * 32 + mi * 16 + half * 8 + r;
                float z = acc[mi][ni][r] + bv;
                float res;
                if (which == 0) res = fmaxf(z, 0.0f) + log1pf(expf(-fabsf(z))); // softplus
                else            res = z / (1.0f + expf(-z));                    // silu
                out[(size_t)row * DMODEL + col] = res;
            }
        }
    }
}

// One block per token row: SSM update + state contraction + gated residual + LayerNorm
__global__ __launch_bounds__(256)
void ssm_scan_ln_kernel(const float* __restrict__ x_t, const float* __restrict__ state,
                        const float* __restrict__ log_A, const float* __restrict__ B,
                        const float* __restrict__ C, const float* __restrict__ gamma,
                        const float* __restrict__ beta,
                        const float* __restrict__ delta, const float* __restrict__ gate,
                        float* __restrict__ y, float* __restrict__ new_state)
{
    const int n   = blockIdx.x;
    const int tid = threadIdx.x;
    __shared__ float red_s[8];
    __shared__ float red_q[8];

    float rvals[4];
    float sum = 0.0f, sq = 0.0f;

#pragma unroll
    for (int i = 0; i < 4; ++i) {
        int d = tid + i * 256;
        size_t nd = (size_t)n * DMODEL + d;
        float xv = x_t[nd];
        float dl = delta[nd];
        float gv = gate[nd];
        float dx = dl * xv;
        float accv = 0.0f;
        size_t sb = nd * SDIM;
        size_t db = (size_t)d * SDIM;
#pragma unroll
        for (int s4 = 0; s4 < 4; ++s4) {
            f4 st = __builtin_nontemporal_load((const f4*)&state[sb + s4 * 4]);
            f4 la = *(const f4*)&log_A[db + s4 * 4];
            f4 bb = *(const f4*)&B[db + s4 * 4];
            f4 cc = *(const f4*)&C[db + s4 * 4];
            f4 ns;
            ns.x = __expf(-dl * __expf(la.x)) * st.x + dx * bb.x;
            ns.y = __expf(-dl * __expf(la.y)) * st.y + dx * bb.y;
            ns.z = __expf(-dl * __expf(la.z)) * st.z + dx * bb.z;
            ns.w = __expf(-dl * __expf(la.w)) * st.w + dx * bb.w;
            accv += ns.x * cc.x + ns.y * cc.y + ns.z * cc.z + ns.w * cc.w;
            __builtin_nontemporal_store(ns, (f4*)&new_state[sb + s4 * 4]);
        }
        float r = xv + gv * accv;
        rvals[i] = r;
        sum += r;
        sq  += r * r;
    }

    // wave32 butterfly reduction, then cross-wave via LDS
#pragma unroll
    for (int m = 16; m >= 1; m >>= 1) {
        sum += __shfl_xor(sum, m, 32);
        sq  += __shfl_xor(sq,  m, 32);
    }
    if ((tid & 31) == 0) { red_s[tid >> 5] = sum; red_q[tid >> 5] = sq; }
    __syncthreads();
    float ts = 0.0f, tq = 0.0f;
#pragma unroll
    for (int w = 0; w < 8; ++w) { ts += red_s[w]; tq += red_q[w]; }

    const float inv = 1.0f / (float)DMODEL;
    float mu   = ts * inv;
    float var  = tq * inv - mu * mu;
    float rstd = rsqrtf(var + EPSLN);

#pragma unroll
    for (int i = 0; i < 4; ++i) {
        int d = tid + i * 256;
        size_t nd = (size_t)n * DMODEL + d;
        float yv = (rvals[i] - mu) * rstd * gamma[d] + beta[d];
        __builtin_nontemporal_store(yv, &y[nd]);
    }
}

extern "C" void kernel_launch(void* const* d_in, const int* in_sizes, int n_in,
                              void* d_out, int out_size, void* d_ws, size_t ws_size,
                              hipStream_t stream) {
    const float* x_t   = (const float*)d_in[0];
    const float* state = (const float*)d_in[1];
    const float* log_A = (const float*)d_in[2];
    const float* B     = (const float*)d_in[3];
    const float* C     = (const float*)d_in[4];
    const float* Wd    = (const float*)d_in[5];
    const float* bd    = (const float*)d_in[6];
    const float* Wg    = (const float*)d_in[7];
    const float* bg    = (const float*)d_in[8];
    const float* gamma = (const float*)d_in[9];
    const float* beta  = (const float*)d_in[10];

    float* y         = (float*)d_out;
    float* new_state = y + (size_t)N_TOK * DMODEL;

    float* delta = (float*)d_ws;                      // N*D f32 = 32 MB
    float* gate  = delta + (size_t)N_TOK * DMODEL;    // N*D f32 = 32 MB

    // Both GEMMs in one launch (z=2) so x stays hot in the 192MB L2.
    dim3 ggrid(N_TOK / 128, DMODEL / 128, 2);
    ssm_gemm_epi_kernel<<<ggrid, 256, 0, stream>>>(x_t, Wd, bd, Wg, bg, delta, gate);

    ssm_scan_ln_kernel<<<N_TOK, 256, 0, stream>>>(x_t, state, log_A, B, C,
                                                  gamma, beta, delta, gate,
                                                  y, new_state);
}
